// PET_15238543966801
// MI455X (gfx1250) — matrix operand order repaired and checked by reading, not compiled, not run on hardware
//
#include <hip/hip_runtime.h>
#include <stdint.h>
#include <math.h>

#define N_NODES 5000
#define N_STRUCT 50
#define PER 100
#define N_EDGES 120000
#define MAXE 32
#define D 128
#define DFF 512
#define HEADS 4
#define NW 2                  // nodes per workgroup (4 waves, 128 threads)
#define USE_TDM 1

// LDS layout (k_block): hbuf/qbuf/kbuf/vbuf 16KB each, 2x 32KB weight panels,
// 4x 3328B per-wave misc, 256B mask  -> 141KB => 2 WGs per WGP (320KB LDS).
#define OFF_HBUF   0
#define OFF_QBUF   16384
#define OFF_KBUF   32768
#define OFF_VBUF   49152
#define OFF_PAN0   65536
#define OFF_PAN1   98304
#define OFF_MISC   131072
#define OFF_MASK   144384
#define SMEM_BLOCK 144640
#define SMEM_CONTRACT 131072  // 64KB g_contr panel + 8*8KB A stages

typedef __bf16 bf16;
typedef __attribute__((ext_vector_type(16))) __bf16 v16bf;
typedef __attribute__((ext_vector_type(8)))  float  v8f;
typedef __attribute__((ext_vector_type(4)))  unsigned int u32x4;
typedef __attribute__((ext_vector_type(8)))  int i32x8;
typedef __attribute__((ext_vector_type(4)))  int i32x4;

#define WMMA_BF16(a,b,c) __builtin_amdgcn_wmma_f32_16x16x32_bf16(false,(a),false,(b),(short)0,(c),false,false)

__device__ inline v8f vzero8(){ v8f z = {0.f,0.f,0.f,0.f,0.f,0.f,0.f,0.f}; return z; }

__device__ inline float gelu_f(float x){
  const float c0 = 0.7978845608028654f;
  float t = tanhf(c0*(x + 0.044715f*x*x*x));
  return 0.5f*x*(1.0f+t);
}

__device__ inline int elem_of(int num){
  return num==1?0: num==6?1: num==7?2: num==8?3 : 0;
}

// ---- WMMA fragment loaders (contiguous k = 16*(lane>>4) + s in A and B, wave32) -------------

// A: 16x32 bf16 tile from row-major bf16 LDS (ld elems), rows = lane&15.
__device__ inline v16bf load_a(const bf16* src, int ld, int kbase, int lane){
  int m = lane & 15, hi = lane >> 4;
  union { v16bf v; uint32_t u[8]; } r;
  const uint32_t* row = (const uint32_t*)(src + (size_t)m*ld);
#pragma unroll
  for (int j=0;j<8;j++){
    int k0 = 2*j + hi*16 + kbase;  // always even
    r.u[j] = row[k0>>1];
  }
  return r.v;
}

// B: 32x16 tile from paired-K packed u32 weight panel: wp[(k/2)*128 + n].
__device__ inline v16bf load_b_pack(const uint32_t* wp, int kbase, int nt, int lane){
  int n = (lane & 15) + nt*16;
  int khalf = (lane >> 4) * 16;
  union { v16bf v; uint32_t u[8]; } r;
#pragma unroll
  for (int j=0;j<8;j++){
    int k0 = kbase + khalf + 2*j;
    r.u[j] = wp[(k0>>1)*128 + n];
  }
  return r.v;
}

// B where B[k][n] = act[n*ld + cbase + k]  (k contiguous -> one u32/VGPR)
__device__ inline v16bf load_b_trans(const bf16* act, int ld, int cbase, int nbase, int lane){
  int n = (lane & 15) + nbase;
  int khalf = (lane >> 4) * 16;
  union { v16bf v; uint32_t u[8]; } r;
  const uint32_t* row = (const uint32_t*)(act + (size_t)n*ld + cbase);
#pragma unroll
  for (int j=0;j<8;j++) r.u[j] = row[(khalf>>1) + j];
  return r.v;
}

// B where B[k][n] = act[k*ld + colbase + n]  (strided 16-bit reads)
__device__ inline v16bf load_b_row(const bf16* act, int ld, int colbase, int lane){
  int col = colbase + (lane & 15);
  int khalf = (lane >> 4) * 16;
  union { v16bf v; uint32_t u[8]; } r;
#pragma unroll
  for (int j=0;j<8;j++){
    int k0 = khalf + 2*j;
    union { bf16 b[2]; uint32_t u; } p;
    p.b[0] = act[(size_t)k0*ld + col];
    p.b[1] = act[(size_t)(k0+1)*ld + col];
    r.u[j] = p.u;
  }
  return r.v;
}

// ---- Tensor Data Mover: 2D tile (w elems x h rows, u32 elems) -> LDS ------
__device__ inline void tdm_load2d(uint32_t lds_off, const void* gaddr,
                                  uint32_t w, uint32_t h, uint32_t stride){
  unsigned long long ga = (unsigned long long)gaddr;
  u32x4 g0;
  g0[0] = 1u;                                         // count=1, user mode
  g0[1] = lds_off;                                    // lds_addr (bytes)
  g0[2] = (uint32_t)(ga & 0xffffffffu);               // global_addr lo
  g0[3] = (uint32_t)((ga >> 32) & 0x01ffffffu) | (2u << 30); // addr hi | type=2
  i32x8 g1;
  g1[0] = (int)(2u << 16);                            // data_size = 4B
  g1[1] = (int)((w & 0xffffu) << 16);                 // tensor_dim0 lo16
  g1[2] = (int)(((w >> 16) & 0xffffu) | ((h & 0xffffu) << 16)); // dim0 hi | dim1 lo
  g1[3] = (int)(((h >> 16) & 0xffffu) | ((w & 0xffffu) << 16)); // dim1 hi | tile_dim0
  g1[4] = (int)(h & 0xffffu);                         // tile_dim1 | tile_dim2=0
  g1[5] = (int)stride;                                // tensor_dim0_stride lo32
  g1[6] = 0;
  g1[7] = 0;
  i32x4 z4 = {0,0,0,0};
#if __clang_major__ >= 23
  i32x8 z8 = {0,0,0,0,0,0,0,0};
  __builtin_amdgcn_tensor_load_to_lds(g0, g1, z4, z4, z8, 0);
#else
  __builtin_amdgcn_tensor_load_to_lds(g0, g1, z4, z4, 0);
#endif
}

// split issue/wait so a panel load can overlap the previous GEMM
__device__ inline void tdm_issue(uint32_t* dst, const uint32_t* src,
                                 int w, int h, int stride){
#if USE_TDM
  if ((threadIdx.x >> 5) == 0)
    tdm_load2d((uint32_t)(uintptr_t)(void*)dst, (const void*)src,
               (uint32_t)w, (uint32_t)h, (uint32_t)stride);
#else
  int tot = w*h;
  for (int t = threadIdx.x; t < tot; t += blockDim.x){
    int r = t / w, c = t - r*w;
    dst[t] = src[(size_t)r*stride + c];
  }
#endif
}
__device__ inline void tdm_wait(){
#if USE_TDM
  if ((threadIdx.x >> 5) == 0) __builtin_amdgcn_s_wait_tensorcnt(0);
#endif
}

// ---- LayerNorm 16 rows x 128 (fp32 global -> bf16 LDS), per wave ----------
__device__ inline void ln_lds(const float* xs, const float* w, const float* b,
                              bf16* dst, float* stats, int lane){
  int r = lane >> 1, half = lane & 1;
  const float* px = xs + (size_t)r*D + half*64;
  float s = 0.f, s2 = 0.f;
#pragma unroll 8
  for (int i=0;i<64;i++){ float v = px[i]; s += v; s2 += v*v; }
  s  += __shfl_xor(s, 1, 32);
  s2 += __shfl_xor(s2, 1, 32);
  float mean = s * (1.f/128.f);
  float var  = s2 * (1.f/128.f) - mean*mean;
  float rstd = rsqrtf(var + 1e-5f);
  if (half == 0){ stats[r*2] = mean; stats[r*2+1] = rstd; }
  for (int t = lane; t < 16*D; t += 32){
    int rr = t >> 7, c = t & 127;
    float v = xs[(size_t)rr*D + c];
    dst[rr*D + c] = (bf16)((v - stats[rr*2]) * stats[rr*2+1] * w[c] + b[c]);
  }
}

// [16,128] = A(16x128 bf16) @ B(128x128 packed), bf16 output rows
__device__ inline void gemm_proj(const bf16* arow, const uint32_t* wp,
                                 bf16* drow, int lane){
  int hi = lane >> 4, ln15 = lane & 15;
  v16bf a4[4];
#pragma unroll
  for (int kc=0;kc<4;kc++) a4[kc] = load_a(arow, D, kc*32, lane);
#pragma unroll
  for (int nt=0;nt<8;nt++){
    v8f c = vzero8();
#pragma unroll
    for (int kc=0;kc<4;kc++) c = WMMA_BF16(a4[kc], load_b_pack(wp, kc*32, nt, lane), c);
    int col = nt*16 + ln15;
#pragma unroll
    for (int r=0;r<8;r++) drow[(r+hi*8)*D + col] = (bf16)c[r];
  }
}

// =========================== kernels =======================================

__global__ void k_pack(const float* __restrict__ src, uint32_t* __restrict__ dst,
                       int Kd, int Nd){
  int idx = blockIdx.x*blockDim.x + threadIdx.x;
  int tot = (Kd >> 1) * Nd;
  if (idx >= tot) return;
  int kk = idx / Nd, n = idx - kk*Nd;
  union { bf16 b[2]; uint32_t u; } p;
  p.b[0] = (bf16)src[(size_t)(2*kk  )*Nd + n];
  p.b[1] = (bf16)src[(size_t)(2*kk+1)*Nd + n];
  dst[idx] = p.u;
}

__global__ void k_precomp(const float* __restrict__ Wcart, const float* __restrict__ embc,
                          const float* __restrict__ embn, const float* __restrict__ Wcomp,
                          float* __restrict__ W3, float* __restrict__ Tc, float* __restrict__ Tn){
  int b = blockIdx.x, d = threadIdx.x;
  float s = 0.f;
  if (b < 3){
    for (int j=0;j<D;j++) s += Wcart[b*D+j] * Wcomp[(size_t)j*D + d];
    W3[b*D+d] = s;
  } else if (b < 7){
    int r = b-3;
    for (int j=0;j<D;j++) s += embc[r*D+j] * Wcomp[(size_t)(D+j)*D + d];
    Tc[r*D+d] = s;
  } else {
    int r = b-7;
    for (int j=0;j<D;j++) s += embn[r*D+j] * Wcomp[(size_t)(2*D+j)*D + d];
    Tn[r*D+d] = s;
  }
}

__global__ void k_graph_start(const int* __restrict__ centers,
                              int* __restrict__ start, int* __restrict__ counts){
  int n = blockIdx.x*blockDim.x + threadIdx.x;
  if (n >= N_NODES) return;
  auto lb = [&](int key){
    int lo = 0, hi = N_EDGES;
    while (lo < hi){ int mid = (lo+hi) >> 1; if (centers[mid] < key) lo = mid+1; else hi = mid; }
    return lo;
  };
  int s0 = lb(n), s1 = lb(n+1);
  start[n] = s0; counts[n] = s1 - s0;
}

__global__ void k_graph_edges(const int* __restrict__ centers, const int* __restrict__ neighbors,
                              const int* __restrict__ start, const int* __restrict__ counts,
                              int* __restrict__ slot, int* __restrict__ nef, int* __restrict__ corr,
                              const float* __restrict__ pos, const float* __restrict__ cells,
                              const float* __restrict__ shifts,
                              float* __restrict__ ev, float* __restrict__ radial){
  int e = blockIdx.x*blockDim.x + threadIdx.x;
  if (e >= N_EDGES) return;
  int c = centers[e];
  int sl = e - start[c];
  slot[e] = sl;
  if (sl < MAXE) nef[c*MAXE + sl] = e;
  int nb = neighbors[e];
  int st = start[nb], cnt = counts[nb];
  int cr = e;
  for (int j=0;j<cnt;j++){ if (neighbors[st+j] == c){ cr = st+j; break; } }
  corr[e] = cr;
  int sg = c / PER;
  float vx[3];
#pragma unroll
  for (int b=0;b<3;b++){
    float v = pos[nb*3+b] - pos[c*3+b];
    for (int a=0;a<3;a++) v += shifts[(size_t)e*3+a] * cells[sg*9 + a*3 + b];
    vx[b] = v; ev[(size_t)e*3+b] = v;
  }
  float r = sqrtf(vx[0]*vx[0] + vx[1]*vx[1] + vx[2]*vx[2]);
  float cut;
  if (r < 3.f) cut = 1.f;
  else if (r >= 5.f) cut = 0.f;
  else cut = 0.5f*(1.f + cosf(3.14159265358979323f*(r-3.f)*0.5f));
  radial[e] = cut;
}

__global__ void k_rnef(const int* __restrict__ counts, const int* __restrict__ nef,
                       const float* __restrict__ radial, float* __restrict__ rnef){
  int t = blockIdx.x*blockDim.x + threadIdx.x;
  if (t >= N_NODES*MAXE) return;
  int n = t >> 5, s = t & 31;
  rnef[t] = (s < counts[n]) ? radial[nef[t]] : 0.f;
}

__global__ void k_feats_init(const int* __restrict__ nef, const int* __restrict__ centers,
                             const int* __restrict__ neighbors, const int* __restrict__ numbers,
                             const float* __restrict__ ev, const float* __restrict__ W3,
                             const float* __restrict__ Tc, const float* __restrict__ Tn,
                             float* __restrict__ feats){
  int idx = blockIdx.x*blockDim.x + threadIdx.x;
  if (idx >= N_NODES*MAXE*D) return;
  int n = idx >> 12;          // /(32*128)
  int rem = idx & 4095;
  int s = rem >> 7, d = rem & 127;
  int e = nef[n*MAXE + s];
  int ct = centers[e], nb = neighbors[e];
  int ec = elem_of(numbers[ct]), en = elem_of(numbers[nb]);
  float x = ev[(size_t)e*3+0], y = ev[(size_t)e*3+1], z = ev[(size_t)e*3+2];
  feats[idx] = x*W3[d] + y*W3[D+d] + z*W3[2*D+d] + Tc[ec*D+d] + Tn[en*D+d];
}

// ---- fused transformer block (LN->QKV->attn->O->LN->MLP), bf16 WMMA -------
// 128 threads = 4 waves, 2 nodes; double-buffered TDM weight panels.
__global__ void __launch_bounds__(128) k_block(
    float* __restrict__ x, const float* __restrict__ mask,
    const uint32_t* __restrict__ wq, const uint32_t* __restrict__ wk,
    const uint32_t* __restrict__ wv, const uint32_t* __restrict__ wo,
    const uint32_t* __restrict__ w1, const uint32_t* __restrict__ w2,
    const float* __restrict__ ln1w, const float* __restrict__ ln1b,
    const float* __restrict__ ln2w, const float* __restrict__ ln2b,
    const float* __restrict__ b1,   const float* __restrict__ b2){
  extern __shared__ char smem[];
  bf16*     hbuf = (bf16*)(smem + OFF_HBUF);   // [2][32][128]
  bf16*     qbuf = (bf16*)(smem + OFF_QBUF);
  bf16*     kbuf = (bf16*)(smem + OFF_KBUF);
  bf16*     vbuf = (bf16*)(smem + OFF_VBUF);   // MLP: per-wave ustage
  uint32_t* pan0 = (uint32_t*)(smem + OFF_PAN0);
  uint32_t* pan1 = (uint32_t*)(smem + OFF_PAN1);
  char*     misc = smem + OFF_MISC;            // per-wave 3328B
  float*   maskb = (float*)(smem + OFF_MASK);

  const int tid = threadIdx.x;
  const int wid = tid >> 5, lane = tid & 31;
  const int hi = lane >> 4, ln15 = lane & 15;
  const int nl = wid >> 1, rt = wid & 1;
  const int node0 = blockIdx.x * NW;
  const int node = node0 + nl;

  float* xrow  = x    + ((size_t)node*MAXE + rt*16) * D;
  bf16*  hrow  = hbuf + (nl*MAXE + rt*16) * D;
  bf16*  qrow  = qbuf + (nl*MAXE + rt*16) * D;
  bf16*  kn    = kbuf + nl*MAXE*D;
  bf16*  vn    = vbuf + nl*MAXE*D;
  float* sc    = (float*)(misc + wid*3328);
  bf16*  pb    = (bf16*) (misc + wid*3328 + 2048);
  float* stats = (float*)(misc + wid*3328 + 3072);

  for (int t = tid; t < NW*MAXE; t += 128) maskb[t] = mask[(size_t)node0*MAXE + t];

  ln_lds(xrow, ln1w, ln1b, hrow, stats, lane);
  tdm_issue(pan0, wq, 128, 64, 128);
  tdm_wait();
  __syncthreads();

  // pipelined QKV: issue next panel, GEMM current, wait, barrier
  tdm_issue(pan1, wk, 128, 64, 128);
  gemm_proj(hrow, pan0, qrow, lane);
  tdm_wait(); __syncthreads();

  tdm_issue(pan0, wv, 128, 64, 128);
  gemm_proj(hrow, pan1, kn + rt*16*D, lane);
  tdm_wait(); __syncthreads();

  tdm_issue(pan1, wo, 128, 64, 128);
  gemm_proj(hrow, pan0, vn + rt*16*D, lane);
  tdm_wait(); __syncthreads();

  // attention: per head scores -> softmax * cutoff-weight -> p@v
  v8f attnC[8];
#pragma unroll
  for (int f=0;f<8;f++) attnC[f] = vzero8();
  const float scale = 0.17677669529663687f;   // 1/sqrt(32)
  for (int h=0; h<HEADS; h++){
    v16bf qa = load_a(qrow, D, h*32, lane);
#pragma unroll
    for (int nt=0; nt<2; nt++){
      v8f c = vzero8();
      c = WMMA_BF16(qa, load_b_trans(kn, D, h*32, nt*16, lane), c);
#pragma unroll
      for (int r=0;r<8;r++) sc[(r+hi*8)*32 + nt*16 + ln15] = c[r]*scale;
    }
    if (lane < 16){
      float mx = -1e30f;
      for (int k2=0;k2<32;k2++) mx = fmaxf(mx, sc[lane*32+k2]);
      float sum = 0.f;
      for (int k2=0;k2<32;k2++) sum += expf(sc[lane*32+k2]-mx);
      float inv = 1.f/sum;
      for (int k2=0;k2<32;k2++)
        pb[lane*32+k2] = (bf16)(expf(sc[lane*32+k2]-mx)*inv*maskb[nl*32+k2]);
    }
    v16bf pa = load_a(pb, 32, 0, lane);
#pragma unroll
    for (int nt=0; nt<2; nt++)
      attnC[2*h+nt] = WMMA_BF16(pa, load_b_row(vn, D, h*32 + nt*16, lane), attnC[2*h+nt]);
  }
#pragma unroll
  for (int f=0;f<8;f++)
    for (int r=0;r<8;r++)
      hrow[(r+hi*8)*D + f*16 + ln15] = (bf16)attnC[f][r];

  // O-projection: Wo already resident in pan1; attn-out/hrow are wave-private rows
  {
    v16bf a4[4];
#pragma unroll
    for (int kc=0;kc<4;kc++) a4[kc] = load_a(hrow, D, kc*32, lane);
#pragma unroll
    for (int nt=0;nt<8;nt++){
      v8f c = vzero8();
#pragma unroll
      for (int kc=0;kc<4;kc++) c = WMMA_BF16(a4[kc], load_b_pack(pan1, kc*32, nt, lane), c);
      int col = nt*16 + ln15;
#pragma unroll
      for (int r=0;r<8;r++) xrow[(r+hi*8)*D + col] += c[r];
    }
  }

  // ---- MLP ----
  ln_lds(xrow, ln2w, ln2b, hrow, stats, lane);
  bf16* ustage = vbuf + wid*16*D;   // vbuf region repurposed per wave
  v16bf ha[4];
#pragma unroll
  for (int kc=0;kc<4;kc++) ha[kc] = load_a(hrow, D, kc*32, lane);
  v8f outC[8];
#pragma unroll
  for (int f=0;f<8;f++) outC[f] = vzero8();
  __syncthreads();                  // all waves done with pan1 (O) and vbuf (attn)
  for (int cc=0; cc<4; cc++){
    tdm_issue(pan0, w1 + cc*128,            128, 64, 512);
    tdm_issue(pan1, w2 + (size_t)cc*64*128, 128, 64, 128);
    tdm_wait();
    __syncthreads();
#pragma unroll
    for (int nt=0;nt<8;nt++){
      v8f c = vzero8();
#pragma unroll
      for (int kc=0;kc<4;kc++) c = WMMA_BF16(ha[kc], load_b_pack(pan0, kc*32, nt, lane), c);
      int col = nt*16 + ln15;
      float bias = b1[cc*128 + col];
#pragma unroll
      for (int r=0;r<8;r++) ustage[(r+hi*8)*D + col] = (bf16)gelu_f(c[r] + bias);
    }
    v16bf ua[4];
#pragma unroll
    for (int kc=0;kc<4;kc++) ua[kc] = load_a(ustage, D, kc*32, lane);
#pragma unroll
    for (int nt=0;nt<8;nt++)
#pragma unroll
      for (int kc=0;kc<4;kc++)
        outC[nt] = WMMA_BF16(ua[kc], load_b_pack(pan1, kc*32, nt, lane), outC[nt]);
    __syncthreads();
  }
#pragma unroll
  for (int nt=0;nt<8;nt++){
    int col = nt*16 + ln15;
#pragma unroll
    for (int r=0;r<8;r++) xrow[(r+hi*8)*D + col] += outC[nt][r] + b2[col];
  }
}

// ---- GNN edge-pair contraction: concat(feat_e, feat_corr) @ g_contr -------
__global__ void __launch_bounds__(256) k_contract(
    const float* __restrict__ feats, float* __restrict__ nfeats,
    const uint32_t* __restrict__ gc, const int* __restrict__ centers,
    const int* __restrict__ slot, const int* __restrict__ corr){
  extern __shared__ char smem[];
  uint32_t* gpan = (uint32_t*)smem;              // [(256/2)][128] u32
  bf16* ast_base = (bf16*)(smem + 64*1024);      // per wave [16][256]
  const int tid = threadIdx.x, wid = tid >> 5, lane = tid & 31;
  tdm_issue(gpan, gc, 128, 128, 128);
  tdm_wait();
  __syncthreads();
  int gw = blockIdx.x*8 + wid;
  if (gw >= N_EDGES/16) return;
  bf16* ast = ast_base + wid*16*256;
  int e0 = gw*16;
  for (int t = lane; t < 16*256; t += 32){
    int row = t >> 8, col = t & 255;
    int e = e0 + row;
    int es = (col < 128) ? e : corr[e];
    int c2 = centers[es], s2 = slot[es];
    int ci = (col < 128) ? col : col - 128;
    ast[row*256 + col] = (bf16)feats[((size_t)c2*MAXE + s2)*D + ci];
  }
  v8f acc[8];
#pragma unroll
  for (int f=0;f<8;f++) acc[f] = vzero8();
  for (int kc=0;kc<8;kc++){
    v16bf a = load_a(ast, 256, kc*32, lane);
#pragma unroll
    for (int nt=0;nt<8;nt++)
      acc[nt] = WMMA_BF16(a, load_b_pack(gpan, kc*32, nt, lane), acc[nt]);
  }
  int hi = lane >> 4, ln15 = lane & 15;
#pragma unroll
  for (int nt=0;nt<8;nt++){
    int col = nt*16 + ln15;
#pragma unroll
    for (int r=0;r<8;r++){
      int e = e0 + r + hi*8;
      nfeats[((size_t)centers[e]*MAXE + slot[e])*D + col] = acc[nt][r];
    }
  }
}

__global__ void k_fixup(float* __restrict__ nfeats, const int* __restrict__ counts,
                        const int* __restrict__ centers, const int* __restrict__ slot){
  int t = blockIdx.x*blockDim.x + threadIdx.x;
  if (t >= N_NODES*MAXE) return;
  int n = t >> 5, s = t & 31;
  int cnt = counts[n]; if (cnt > MAXE) cnt = MAXE;
  if (s < cnt) return;
  const float* src = nfeats + ((size_t)centers[0]*MAXE + slot[0])*D;  // tmp[edge 0]
  float* dst = nfeats + (size_t)t*D;
  for (int d=0; d<D; d++) dst[d] = src[d];
}

__global__ void k_add(float* __restrict__ a, const float* __restrict__ b){
  int idx = blockIdx.x*blockDim.x + threadIdx.x;
  if (idx >= N_NODES*MAXE*D) return;
  a[idx] += b[idx];
}

__global__ void k_readout(const float* __restrict__ feats, const float* __restrict__ rnef,
                          const float* __restrict__ rW, const float* __restrict__ compW,
                          const int* __restrict__ numbers, float* __restrict__ out){
  int s = blockIdx.x, tid = threadIdx.x;
  __shared__ float red[256];
  float acc = 0.f;
  int tok0 = s * PER * MAXE;
  for (int t = tid; t < PER*MAXE; t += 256){
    float m = rnef[tok0 + t];
    if (m != 0.f){
      const float* f = feats + (size_t)(tok0 + t)*D;
      float d = 0.f;
      for (int i=0;i<D;i++) d += f[i]*rW[i];
      acc += d*m;
    }
  }
  for (int n = tid; n < PER; n += 256)
    acc += compW[elem_of(numbers[s*PER + n])];
  red[tid] = acc; __syncthreads();
  for (int st = 128; st > 0; st >>= 1){
    if (tid < st) red[tid] += red[tid+st];
    __syncthreads();
  }
  if (tid == 0) out[s] = red[0];
}

// =========================== host ==========================================

extern "C" void kernel_launch(void* const* d_in, const int* in_sizes, int n_in,
                              void* d_out, int out_size, void* d_ws, size_t ws_size,
                              hipStream_t stream){
  (void)in_sizes; (void)n_in; (void)out_size; (void)ws_size;
  const float* positions=(const float*)d_in[0];
  const float* cells    =(const float*)d_in[1];
  const float* shifts   =(const float*)d_in[2];
  const float* Wcart    =(const float*)d_in[3];
  const float* embc     =(const float*)d_in[4];
  const float* embn     =(const float*)d_in[5];
  const float* Wcomp    =(const float*)d_in[6];
  const float* tWq=(const float*)d_in[7],  *tWk=(const float*)d_in[8];
  const float* tWv=(const float*)d_in[9],  *tWo=(const float*)d_in[10];
  const float* tln1w=(const float*)d_in[11], *tln1b=(const float*)d_in[12];
  const float* tln2w=(const float*)d_in[13], *tln2b=(const float*)d_in[14];
  const float* tW1=(const float*)d_in[15], *tb1=(const float*)d_in[16];
  const float* tW2=(const float*)d_in[17], *tb2=(const float*)d_in[18];
  const float* gcon=(const float*)d_in[19];
  const float* gWq=(const float*)d_in[20], *gWk=(const float*)d_in[21];
  const float* gWv=(const float*)d_in[22], *gWo=(const float*)d_in[23];
  const float* gln1w=(const float*)d_in[24], *gln1b=(const float*)d_in[25];
  const float* gln2w=(const float*)d_in[26], *gln2b=(const float*)d_in[27];
  const float* gW1=(const float*)d_in[28], *gb1=(const float*)d_in[29];
  const float* gW2=(const float*)d_in[30], *gb2=(const float*)d_in[31];
  const float* rW=(const float*)d_in[32];
  const float* compW=(const float*)d_in[33];
  const int* numbers=(const int*)d_in[34];
  const int* centers=(const int*)d_in[35];
  const int* neighbors=(const int*)d_in[36];
  float* out=(float*)d_out;

  char* w = (char*)d_ws; size_t off = 0;
  auto A = [&](size_t bytes)->char*{
    char* p = w + off; off = (off + bytes + 255) & ~(size_t)255; return p;
  };
  int*   start =(int*)A((size_t)N_NODES*4);
  int*   counts=(int*)A((size_t)N_NODES*4);
  int*   nef   =(int*)A((size_t)N_NODES*MAXE*4);
  int*   slot  =(int*)A((size_t)N_EDGES*4);
  int*   corr  =(int*)A((size_t)N_EDGES*4);
  float* ev    =(float*)A((size_t)N_EDGES*3*4);
  float* radial=(float*)A((size_t)N_EDGES*4);
  float* rnef  =(float*)A((size_t)N_NODES*MAXE*4);
  float* feats =(float*)A((size_t)N_NODES*MAXE*D*4);
  float* nfeats=(float*)A((size_t)N_NODES*MAXE*D*4);
  float* W3=(float*)A(3*D*4); float* Tc=(float*)A(4*D*4); float* Tn=(float*)A(4*D*4);
  uint32_t* pk[4][6];
  for (int b=0;b<4;b++){
    for (int i=0;i<4;i++) pk[b][i]=(uint32_t*)A(32768);
    pk[b][4]=(uint32_t*)A(131072);
    pk[b][5]=(uint32_t*)A(131072);
  }
  uint32_t* gcp=(uint32_t*)A(65536);

  // pack weights to paired-K bf16
  for (int b=0;b<4;b++){
    int l = (b < 2) ? b : (b-2);
    const float *sq,*sk,*sv,*so,*s1,*s2;
    if (b < 2){ sq=tWq+(size_t)l*D*D; sk=tWk+(size_t)l*D*D; sv=tWv+(size_t)l*D*D; so=tWo+(size_t)l*D*D;
                s1=tW1+(size_t)l*D*DFF; s2=tW2+(size_t)l*DFF*D; }
    else      { sq=gWq+(size_t)l*D*D; sk=gWk+(size_t)l*D*D; sv=gWv+(size_t)l*D*D; so=gWo+(size_t)l*D*D;
                s1=gW1+(size_t)l*D*DFF; s2=gW2+(size_t)l*DFF*D; }
    k_pack<<<32,256,0,stream>>>(sq, pk[b][0], D, D);
    k_pack<<<32,256,0,stream>>>(sk, pk[b][1], D, D);
    k_pack<<<32,256,0,stream>>>(sv, pk[b][2], D, D);
    k_pack<<<32,256,0,stream>>>(so, pk[b][3], D, D);
    k_pack<<<128,256,0,stream>>>(s1, pk[b][4], D, DFF);
    k_pack<<<128,256,0,stream>>>(s2, pk[b][5], DFF, D);
  }
  k_pack<<<64,256,0,stream>>>(gcon, gcp, 2*D, D);
  k_precomp<<<11,128,0,stream>>>(Wcart, embc, embn, Wcomp, W3, Tc, Tn);

  hipMemsetAsync(nef, 0, (size_t)N_NODES*MAXE*4, stream);
  k_graph_start<<<(N_NODES+255)/256,256,0,stream>>>(centers, start, counts);
  k_graph_edges<<<(N_EDGES+255)/256,256,0,stream>>>(centers,neighbors,start,counts,
      slot,nef,corr,positions,cells,shifts,ev,radial);
  k_rnef<<<(N_NODES*MAXE+255)/256,256,0,stream>>>(counts,nef,radial,rnef);
  k_feats_init<<<(N_NODES*MAXE*D+255)/256,256,0,stream>>>(nef,centers,neighbors,numbers,
      ev,W3,Tc,Tn,feats);

  hipFuncSetAttribute((const void*)k_block, hipFuncAttributeMaxDynamicSharedMemorySize, SMEM_BLOCK);
  hipFuncSetAttribute((const void*)k_contract, hipFuncAttributeMaxDynamicSharedMemorySize, SMEM_CONTRACT);

  for (int b=0;b<2;b++)
    k_block<<<N_NODES/NW,128,SMEM_BLOCK,stream>>>(feats, rnef,
        pk[b][0],pk[b][1],pk[b][2],pk[b][3],pk[b][4],pk[b][5],
        tln1w+b*D, tln1b+b*D, tln2w+b*D, tln2b+b*D, tb1+(size_t)b*DFF, tb2+b*D);

  k_contract<<<(N_EDGES/16+7)/8,256,SMEM_CONTRACT,stream>>>(feats,nfeats,gcp,centers,slot,corr);
  k_fixup<<<(N_NODES*MAXE+255)/256,256,0,stream>>>(nfeats,counts,centers,slot);

  for (int b=2;b<4;b++){
    int l = b-2;
    k_block<<<N_NODES/NW,128,SMEM_BLOCK,stream>>>(nfeats, rnef,
        pk[b][0],pk[b][1],pk[b][2],pk[b][3],pk[b][4],pk[b][5],
        gln1w+l*D, gln1b+l*D, gln2w+l*D, gln2b+l*D, gb1+(size_t)l*DFF, gb2+l*D);
  }
  k_add<<<(N_NODES*MAXE*D+255)/256,256,0,stream>>>(feats,nfeats);
  k_readout<<<N_STRUCT,256,0,stream>>>(feats,rnef,rW,compW,numbers,out);
}
